// SpatialBlock_4904852652760
// MI455X (gfx1250) — matrix operand order repaired and checked by reading, not compiled, not run on hardware
//
#include <hip/hip_runtime.h>
#include <hip/hip_bf16.h>

typedef __attribute__((ext_vector_type(16))) _Float16 v16h;
typedef __attribute__((ext_vector_type(8)))  _Float16 v8h;
typedef __attribute__((ext_vector_type(8)))  float    v8f;

#define HH 112
#define WW 112
#define WS 7
#define HEADS 8
#define DIM 512
#define HID 2048
#define BATCH 4
#define LTOK (HH*WW)             // 12544
#define NTOK (BATCH*LTOK)        // 50176 tokens total (= nW*49)
#define NWIN (BATCH*16*16)       // 1024 windows
#define NPOS 49

union V16U { v16h v; v8h h[2]; };

// ---- WMMA fragment loads (CDNA5 16-bit layouts) ----
// A (16x32, f16): lane = 16*(hi)+row; lane<16 holds K = kk+[0..7] and kk+[16..23],
// lane>=16 holds K = kk+[8..15] and kk+[24..31]  -> two 16B loads per lane.
__device__ __forceinline__ v16h load_a_frag(const _Float16* A, int ldk, int row, int kk, int lane) {
  const _Float16* p = A + (size_t)row * ldk + kk + ((lane >> 4) << 3);
  V16U r;
  r.h[0] = *(const v8h*)(p);
  r.h[1] = *(const v8h*)(p + 16);
  return r.v;
}
// B (32x16, f16, row-major [K][N]): lane l holds column N=n0+(l&15); lane<16 holds K = kk+[0..7] and kk+[16..23],
// lane>=16 holds K = kk+[8..15] and kk+[24..31] (same order as A).
__device__ __forceinline__ v16h load_b_frag(const _Float16* B, int ldn, int kk, int n0, int lane) {
  const _Float16* p = B + (size_t)(kk + ((lane >> 4) << 3)) * ldn + n0 + (lane & 15);
  v16h r;
#pragma unroll
  for (int s = 0; s < 8; ++s) {
    r[s] = p[(size_t)s * ldn];
    r[s + 8] = p[(size_t)(s + 16) * ldn];
  }
  return r;
}

// One wave computes a 16(M) x 64(N) strip: acc[4] of v8f, K stepped by 32.
__device__ __forceinline__ void gemm_strip(const _Float16* A, const _Float16* B,
                                           int K, int N, int m0, int n0, int lane,
                                           v8f acc[4]) {
  int row = m0 + (lane & 15);
  for (int kk = 0; kk < K; kk += 32) {
    v16h a = load_a_frag(A, K, row, kk, lane);
#pragma unroll
    for (int nt = 0; nt < 4; ++nt) {
      v16h b = load_b_frag(B, N, kk, n0 + nt * 16, lane);
      acc[nt] = __builtin_amdgcn_wmma_f32_16x16x32_f16(
          false, a, false, b, (short)0, acc[nt], false, false);
    }
  }
}

// ---------------- elementwise / prep kernels ----------------

__global__ void cvt_f32_f16_kernel(const float* __restrict__ in,
                                   _Float16* __restrict__ out, int n) {
  int i = blockIdx.x * 256 + threadIdx.x;
  if (i < n) out[i] = (_Float16)in[i];
}

// out = x + dwconv3x3(x, w) + b   (NHWC, depthwise, SAME)
__global__ void cpe_kernel(const float* __restrict__ x, const float* __restrict__ w,
                           const float* __restrict__ b, float* __restrict__ out) {
  int idx = blockIdx.x * 256 + threadIdx.x;       // over NTOK*DIM
  int c = idx & (DIM - 1);
  int t = idx >> 9;
  int bi = t / LTOK;
  int rem = t - bi * LTOK;
  int y = rem / WW;
  int xx = rem - y * WW;
  float acc = 0.f;
#pragma unroll
  for (int ky = 0; ky < 3; ++ky) {
    int iy = y + ky - 1;
    if (iy < 0 || iy >= HH) continue;
#pragma unroll
    for (int kx = 0; kx < 3; ++kx) {
      int ix = xx + kx - 1;
      if (ix < 0 || ix >= WW) continue;
      acc += x[(((size_t)bi * LTOK + iy * WW + ix) << 9) + c] * w[((ky * 3 + kx) << 9) + c];
    }
  }
  out[idx] = x[idx] + acc + b[c];
}

// LayerNorm over C=512 per token; f16 output, optionally in window-partition order.
__global__ void ln_kernel(const float* __restrict__ in, const float* __restrict__ g,
                          const float* __restrict__ bb, _Float16* __restrict__ out,
                          int window_order) {
  int t = blockIdx.x;
  int tid = threadIdx.x;
  const float* xp = in + (size_t)t * DIM;
  float a0 = xp[tid], a1 = xp[tid + 256];
  __shared__ float s1[256], s2[256];
  s1[tid] = a0 + a1;
  s2[tid] = a0 * a0 + a1 * a1;
  __syncthreads();
  for (int off = 128; off > 0; off >>= 1) {
    if (tid < off) { s1[tid] += s1[tid + off]; s2[tid] += s2[tid + off]; }
    __syncthreads();
  }
  float mu = s1[0] * (1.f / DIM);
  float var = s2[0] * (1.f / DIM) - mu * mu;
  float rinv = rsqrtf(var + 1e-5f);
  size_t obase;
  if (window_order) {
    int bi = t / LTOK;
    int rem = t - bi * LTOK;
    int y = rem / WW, xx = rem - (rem / WW) * WW;
    int win = bi * 256 + (y / WS) * 16 + (xx / WS);
    int pos = (y % WS) * WS + (xx % WS);
    obase = ((size_t)win * NPOS + pos) * DIM;
  } else {
    obase = (size_t)t * DIM;
  }
  out[obase + tid]       = (_Float16)((a0 - mu) * rinv * g[tid] + bb[tid]);
  out[obase + tid + 256] = (_Float16)((a1 - mu) * rinv * g[tid + 256] + bb[tid + 256]);
}

// ---------------- WMMA GEMM kernels (256 thr = 8 waves; tile 128x64) ----------------

// QKV: [50176,512] x [512,1536]; scatter into q (scaled), kT (transposed), v.
__global__ void gemm_qkv_kernel(const _Float16* __restrict__ A, const _Float16* __restrict__ B,
                                const float* __restrict__ bias,
                                _Float16* __restrict__ qb, _Float16* __restrict__ ktb,
                                _Float16* __restrict__ vb) {
  const int K = DIM, N = 3 * DIM;
  int lane = threadIdx.x & 31, wv = threadIdx.x >> 5;
  int m0 = blockIdx.x * 128 + wv * 16;
  int n0 = blockIdx.y * 64;
  v8f acc[4] = {};
  gemm_strip(A, B, K, N, m0, n0, lane, acc);
#pragma unroll
  for (int nt = 0; nt < 4; ++nt)
#pragma unroll
    for (int r = 0; r < 8; ++r) {
      int m = m0 + r + ((lane >> 4) << 3);
      int n = n0 + nt * 16 + (lane & 15);
      float val = acc[nt][r] + bias[n];
      int win = m / NPOS, pos = m - win * NPOS;
      int which = n >> 9, head = (n >> 6) & 7, dd = n & 63;
      size_t base = ((size_t)(win * HEADS + head)) << 12;  // *64*64
      if (which == 0)      qb[base + pos * 64 + dd] = (_Float16)(val * 0.125f); // *d^-0.5
      else if (which == 1) ktb[base + dd * 64 + pos] = (_Float16)val;
      else                 vb[base + pos * 64 + dd] = (_Float16)val;
    }
}

// Proj: [50176,512] x [512,512]; fuse window-reverse + shortcut residual -> f32 h.
__global__ void gemm_proj_kernel(const _Float16* __restrict__ A, const _Float16* __restrict__ B,
                                 const float* __restrict__ bias,
                                 const float* __restrict__ shortcut, float* __restrict__ hout) {
  const int K = DIM, N = DIM;
  int lane = threadIdx.x & 31, wv = threadIdx.x >> 5;
  int m0 = blockIdx.x * 128 + wv * 16;
  int n0 = blockIdx.y * 64;
  v8f acc[4] = {};
  gemm_strip(A, B, K, N, m0, n0, lane, acc);
#pragma unroll
  for (int nt = 0; nt < 4; ++nt)
#pragma unroll
    for (int r = 0; r < 8; ++r) {
      int m = m0 + r + ((lane >> 4) << 3);
      int n = n0 + nt * 16 + (lane & 15);
      int win = m / NPOS, pos = m - win * NPOS;
      int bi = win >> 8, wy = (win >> 4) & 15, wx = win & 15;
      int y = wy * WS + pos / WS, xx = wx * WS + pos % WS;
      size_t o = (((size_t)bi * LTOK + y * WW + xx) << 9) + n;
      hout[o] = shortcut[o] + acc[nt][r] + bias[n];
    }
}

// FC1 (chunk of 12544 rows): [M,512] x [512,2048]; fuse tanh-GELU -> f16.
__global__ void gemm_fc1_kernel(const _Float16* __restrict__ A, const _Float16* __restrict__ B,
                                const float* __restrict__ bias, _Float16* __restrict__ out16) {
  const int K = DIM, N = HID;
  int lane = threadIdx.x & 31, wv = threadIdx.x >> 5;
  int m0 = blockIdx.x * 128 + wv * 16;
  int n0 = blockIdx.y * 64;
  v8f acc[4] = {};
  gemm_strip(A, B, K, N, m0, n0, lane, acc);
#pragma unroll
  for (int nt = 0; nt < 4; ++nt)
#pragma unroll
    for (int r = 0; r < 8; ++r) {
      int m = m0 + r + ((lane >> 4) << 3);
      int n = n0 + nt * 16 + (lane & 15);
      float u = acc[nt][r] + bias[n];
      float gl = 0.5f * u * (1.f + tanhf(0.7978845608f * (u + 0.044715f * u * u * u)));
      out16[(size_t)m * HID + n] = (_Float16)gl;
    }
}

// FC2 (chunk): [M,2048] x [2048,512]; fuse residual with h2 -> f32 final out.
__global__ void gemm_fc2_kernel(const _Float16* __restrict__ A, const _Float16* __restrict__ B,
                                const float* __restrict__ bias,
                                const float* __restrict__ h2, float* __restrict__ out) {
  const int K = HID, N = DIM;
  int lane = threadIdx.x & 31, wv = threadIdx.x >> 5;
  int m0 = blockIdx.x * 128 + wv * 16;
  int n0 = blockIdx.y * 64;
  v8f acc[4] = {};
  gemm_strip(A, B, K, N, m0, n0, lane, acc);
#pragma unroll
  for (int nt = 0; nt < 4; ++nt)
#pragma unroll
    for (int r = 0; r < 8; ++r) {
      int m = m0 + r + ((lane >> 4) << 3);
      int n = n0 + nt * 16 + (lane & 15);
      size_t o = (size_t)m * DIM + n;
      out[o] = h2[o] + acc[nt][r] + bias[n];
    }
}

// ---------------- windowed attention (1 wave = 1 head; 2 heads/block) ----------------
__global__ void attn_kernel(const _Float16* __restrict__ q, const _Float16* __restrict__ kT,
                            const _Float16* __restrict__ v, _Float16* __restrict__ out) {
  __shared__ __attribute__((aligned(16))) float    Ssm[2][64 * 64];
  __shared__ __attribute__((aligned(16))) _Float16 Psm[2][64 * 64];
  int wv = threadIdx.x >> 5, lane = threadIdx.x & 31;
  int win = blockIdx.x >> 2, hp = blockIdx.x & 3;
  int head = hp * 2 + wv;
  size_t hb = ((size_t)(win * HEADS + head)) << 12;
  const _Float16* qh = q + hb;
  const _Float16* kh = kT + hb;
  const _Float16* vh = v + hb;
  float* S = Ssm[wv];
  _Float16* P = Psm[wv];

  // S = (q*scale) @ k^T   (64x64 padded; valid 49x49)
  for (int mt = 0; mt < 4; ++mt) {
    v8f acc[4] = {};
    int row = mt * 16 + (lane & 15);
    for (int kk = 0; kk < 64; kk += 32) {
      v16h a = load_a_frag(qh, 64, row, kk, lane);
#pragma unroll
      for (int nt = 0; nt < 4; ++nt) {
        v16h b = load_b_frag(kh, 64, kk, nt * 16, lane);
        acc[nt] = __builtin_amdgcn_wmma_f32_16x16x32_f16(
            false, a, false, b, (short)0, acc[nt], false, false);
      }
    }
#pragma unroll
    for (int nt = 0; nt < 4; ++nt)
#pragma unroll
      for (int r = 0; r < 8; ++r)
        S[(mt * 16 + r + ((lane >> 4) << 3)) * 64 + nt * 16 + (lane & 15)] = acc[nt][r];
  }
  __syncthreads();

  // masked softmax over the 49 valid keys; zero pad rows/cols of P
  for (int row = lane; row < 64; row += 32) {
    if (row < NPOS) {
      float mx = -1e30f;
      for (int c = 0; c < NPOS; ++c) mx = fmaxf(mx, S[row * 64 + c]);
      float sum = 0.f;
      for (int c = 0; c < NPOS; ++c) {
        float e = __expf(S[row * 64 + c] - mx);
        sum += e;
        S[row * 64 + c] = e;
      }
      float inv = 1.f / sum;
      for (int c = 0; c < 64; ++c)
        P[row * 64 + c] = (c < NPOS) ? (_Float16)(S[row * 64 + c] * inv) : (_Float16)0.f;
    } else {
      for (int c = 0; c < 64; ++c) P[row * 64 + c] = (_Float16)0.f;
    }
  }
  __syncthreads();

  // O = P @ V  (pad cols of P are zero, so padded V tokens contribute nothing)
  for (int mt = 0; mt < 4; ++mt) {
    v8f acc[4] = {};
    int row = mt * 16 + (lane & 15);
    for (int kk = 0; kk < 64; kk += 32) {
      v16h a = load_a_frag(P, 64, row, kk, lane);
#pragma unroll
      for (int nt = 0; nt < 4; ++nt) {
        v16h b = load_b_frag(vh, 64, kk, nt * 16, lane);
        acc[nt] = __builtin_amdgcn_wmma_f32_16x16x32_f16(
            false, a, false, b, (short)0, acc[nt], false, false);
      }
    }
#pragma unroll
    for (int nt = 0; nt < 4; ++nt)
#pragma unroll
      for (int r = 0; r < 8; ++r) {
        int m = mt * 16 + r + ((lane >> 4) << 3);
        if (m < NPOS) {
          size_t token = (size_t)win * NPOS + m;
          out[token * DIM + head * 64 + nt * 16 + (lane & 15)] = (_Float16)acc[nt][r];
        }
      }
  }
}

// ---------------- host orchestration ----------------

extern "C" void kernel_launch(void* const* d_in, const int* in_sizes, int n_in,
                              void* d_out, int out_size, void* d_ws, size_t ws_size,
                              hipStream_t stream) {
  const float* x          = (const float*)d_in[0];
  const float* cpe0_w     = (const float*)d_in[1];
  const float* cpe0_b     = (const float*)d_in[2];
  const float* cpe1_w     = (const float*)d_in[3];
  const float* cpe1_b     = (const float*)d_in[4];
  const float* ln1_g      = (const float*)d_in[5];
  const float* ln1_b      = (const float*)d_in[6];
  const float* ln2_g      = (const float*)d_in[7];
  const float* ln2_b      = (const float*)d_in[8];
  const float* qkv_w      = (const float*)d_in[9];
  const float* qkv_b      = (const float*)d_in[10];
  const float* proj_w     = (const float*)d_in[11];
  const float* proj_b     = (const float*)d_in[12];
  const float* fc1_w      = (const float*)d_in[13];
  const float* fc1_b      = (const float*)d_in[14];
  const float* fc2_w      = (const float*)d_in[15];
  const float* fc2_b      = (const float*)d_in[16];
  float* out = (float*)d_out;

  const size_t S1 = (size_t)NTOK * DIM * 4;       // f32 token buffer
  const size_t S2 = (size_t)NTOK * DIM * 2;       // f16 token buffer
  const size_t QS = (size_t)NWIN * HEADS * 64 * 64 * 2;

  char* w = (char*)d_ws;
  float*    shortcut = (float*)(w);
  float*    hbuf     = (float*)(w + S1);
  float*    h2buf    = (float*)(w + 2 * S1);
  _Float16* xw16     = (_Float16*)(w + 3 * S1);            // LN1 (window order); later LN2
  _Float16* qbuf     = (_Float16*)(w + 3 * S1 + S2);       // later: fc1 chunk output
  _Float16* ktbuf    = (_Float16*)(w + 3 * S1 + S2 + QS);
  _Float16* vbuf     = (_Float16*)(w + 3 * S1 + S2 + 2 * QS);
  _Float16* attn16   = (_Float16*)(w + 3 * S1 + S2 + 3 * QS);
  char* wt = w + 3 * S1 + 2 * S2 + 3 * QS;
  _Float16* qkvw16 = (_Float16*)(wt);
  _Float16* projw16 = (_Float16*)(wt + (size_t)DIM * 3 * DIM * 2);
  _Float16* fc1w16 = (_Float16*)(wt + (size_t)DIM * 3 * DIM * 2 + (size_t)DIM * DIM * 2);
  _Float16* fc2w16 = (_Float16*)(wt + (size_t)DIM * 3 * DIM * 2 + (size_t)DIM * DIM * 2 +
                                 (size_t)DIM * HID * 2);
  _Float16* ln216 = xw16;      // reuse (xw16 dead after QKV GEMM)
  _Float16* fc1c  = qbuf;      // reuse q/kT/v region (dead after attention)

  // weight conversion f32 -> f16
  cvt_f32_f16_kernel<<<(DIM * 3 * DIM + 255) / 256, 256, 0, stream>>>(qkv_w, qkvw16, DIM * 3 * DIM);
  cvt_f32_f16_kernel<<<(DIM * DIM + 255) / 256, 256, 0, stream>>>(proj_w, projw16, DIM * DIM);
  cvt_f32_f16_kernel<<<(DIM * HID + 255) / 256, 256, 0, stream>>>(fc1_w, fc1w16, DIM * HID);
  cvt_f32_f16_kernel<<<(HID * DIM + 255) / 256, 256, 0, stream>>>(fc2_w, fc2w16, HID * DIM);

  // shortcut = x + CPE0(x)
  cpe_kernel<<<(NTOK * DIM) / 256, 256, 0, stream>>>(x, cpe0_w, cpe0_b, shortcut);
  // LN1 -> f16 in window-partition order
  ln_kernel<<<NTOK, 256, 0, stream>>>(shortcut, ln1_g, ln1_b, xw16, 1);
  // QKV GEMM, scatter into q/kT/v
  gemm_qkv_kernel<<<dim3(NTOK / 128, (3 * DIM) / 64), 256, 0, stream>>>(
      xw16, qkvw16, qkv_b, qbuf, ktbuf, vbuf);
  // windowed attention
  attn_kernel<<<NWIN * 4, 64, 0, stream>>>(qbuf, ktbuf, vbuf, attn16);
  // proj GEMM + window reverse + residual -> h
  gemm_proj_kernel<<<dim3(NTOK / 128, DIM / 64), 256, 0, stream>>>(
      attn16, projw16, proj_b, shortcut, hbuf);
  // h2 = h + CPE1(h)
  cpe_kernel<<<(NTOK * DIM) / 256, 256, 0, stream>>>(hbuf, cpe1_w, cpe1_b, h2buf);
  // LN2 -> f16 (linear order)
  ln_kernel<<<NTOK, 256, 0, stream>>>(h2buf, ln2_g, ln2_b, ln216, 0);
  // MLP in 4 M-chunks of 12544 rows (caps GELU intermediate at 51 MB)
  for (int ch = 0; ch < 4; ++ch) {
    size_t rowoff = (size_t)ch * LTOK;
    gemm_fc1_kernel<<<dim3(LTOK / 128, HID / 64), 256, 0, stream>>>(
        ln216 + rowoff * DIM, fc1w16, fc1_b, fc1c);
    gemm_fc2_kernel<<<dim3(LTOK / 128, DIM / 64), 256, 0, stream>>>(
        fc1c, fc2w16, fc2_b, h2buf + rowoff * DIM, out + rowoff * DIM);
  }
  (void)in_sizes; (void)n_in; (void)out_size; (void)ws_size;
}
